// FovealBlock_17042430230915
// MI455X (gfx1250) — matrix operand order repaired and checked by reading, not compiled, not run on hardware
//
#include <hip/hip_runtime.h>

typedef __attribute__((ext_vector_type(16))) _Float16 v16h;
typedef __attribute__((ext_vector_type(8)))  float    v8f;

#define K7      7
#define IMGSZ   224
#define LSZ     112
#define CIN     3
#define COUT    64
#define BATCH   32
#define PADSZ   12
#define KDIM    147           // CIN * 7 * 7
#define KSTEPS  5             // ceil(147 / 32)
#define NPQ     49
#define PIX     (LSZ * LSZ)   // 12544
#define MTOT    (BATCH * PIX) // 401408
#define EPSV    1e-5f

// ---------------------------------------------------------------------------
// Kernel 1: fused gather + im2col-in-LDS + GEMM (WMMA f16->f32) + stats
// One wave computes a 64-channel x 16-pixel tile. Block = 8 waves = 128 pixels.
// ---------------------------------------------------------------------------
__global__ __launch_bounds__(256) void foveal_gemm(
    const float* __restrict__ x,       // (B, CIN, 224, 224)
    const float* __restrict__ w,       // (COUT, 147)
    const int*   __restrict__ locs_i,  // (112,112,7,7)
    const int*   __restrict__ locs_j,  // (112,112,7,7)
    float*       __restrict__ out,     // (B, COUT, 112, 112)
    float*       __restrict__ stats)   // [0..63]=sum, [64..127]=sumsq
{
  __shared__ v16h  aw_v[KSTEPS * 4 * 32];     // weight A-fragments   (20 KB)
  __shared__ v16h  bp_v[8 * KSTEPS * 32];     // im2col B-fragments   (40 KB)
  __shared__ int   s_poff[8 * 16 * NPQ];      // per-pixel offsets    (24.5 KB)
  __shared__ float s_sum[COUT];
  __shared__ float s_sq[COUT];

  const int tid  = threadIdx.x;
  const int lane = tid & 31;
  const int wave = tid >> 5;

  if (tid < COUT) { s_sum[tid] = 0.f; s_sq[tid] = 0.f; }

  // ---- weight A-fragments: lane l holds row M=l&15 ; e -> K = e+8*[e>=8]+8*(l>>4)
  {
    _Float16* awh = (_Float16*)aw_v;
    for (int idx = tid; idx < KSTEPS * 4 * 32 * 16; idx += 256) {
      const int e  = idx & 15;
      const int ln = (idx >> 4) & 31;
      const int t  = (idx >> 9) & 3;
      const int ks = idx >> 11;
      const int kl = e + (e >= 8 ? 8 : 0) + ((ln >> 4) ? 8 : 0);
      const int k  = ks * 32 + kl;
      const int ch = t * 16 + (ln & 15);
      _Float16 v = (_Float16)0.f;
      if (k < KDIM) v = (_Float16)w[ch * KDIM + k];
      awh[idx] = v;
    }
  }

  // ---- wave tile position (16 pixels share b,i; j0..j0+15 contiguous) ----
  const int tile = blockIdx.x * 8 + wave;
  const int m0   = tile * 16;
  const int b0   = m0 / PIX;
  const int r0   = m0 - b0 * PIX;
  const int i0   = r0 / LSZ;
  const int j0   = r0 - i0 * LSZ;

  // ---- Stage A: coalesced locs load -> branch-free per-pixel offsets ----
  {
    const long lbase = (long)(i0 * LSZ + j0) * NPQ;   // contiguous 784 ints
    for (int idx = lane; idx < 16 * NPQ; idx += 32) {
      const int ii = locs_i[lbase + idx] - PADSZ;
      const int jj = locs_j[lbase + idx] - PADSZ;
      const bool inb = ((unsigned)ii < (unsigned)IMGSZ) &&
                       ((unsigned)jj < (unsigned)IMGSZ);
      s_poff[wave * (16 * NPQ) + idx] = inb ? (ii * IMGSZ + jj) : -1;
    }
  }
  __syncthreads();

  // ---- Stage B: batched unconditional gathers -> f16 B-fragments in LDS ----
  {
    const int  ppix = tid >> 1;                 // local pixel 0..127
    const int  wv   = ppix >> 4;
    const int  pl   = ppix & 15;
    const int  kh   = tid & 1;                  // k half: 0..79 / 80..159
    const int  mg   = blockIdx.x * 128 + ppix;
    const int  bb   = mg / PIX;
    const long xbb  = (long)bb * (CIN * IMGSZ * IMGSZ);
    _Float16*  bph  = (_Float16*)bp_v;
    const int  pob  = wv * (16 * NPQ) + pl * NPQ;

    #pragma unroll
    for (int chk = 0; chk < 5; ++chk) {
      int addr[16], pad[16];
      #pragma unroll
      for (int u = 0; u < 16; ++u) {
        const int k  = kh * 80 + chk * 16 + u;
        const int kk = (k < KDIM) ? k : (KDIM - 1);   // tail -> zero weights
        const int c  = (kk >= 49) + (kk >= 98);
        const int pq = kk - c * 49;
        const int po = s_poff[pob + pq];
        pad[u]  = (po < 0);
        addr[u] = (po < 0 ? 0 : po) + c * (IMGSZ * IMGSZ);
      }
      float xv[16];
      #pragma unroll
      for (int u = 0; u < 16; ++u) xv[u] = x[xbb + addr[u]];   // batched loads
      #pragma unroll
      for (int u = 0; u < 16; ++u) {
        const int k  = kh * 80 + chk * 16 + u;
        const int ks = k >> 5;
        const int kr = k & 31;
        const int lb = ((kr >> 3) & 1) * 16 + pl;     // B lane: col=pixel, half=(K>>3)&1
        const int e  = (kr & 7) + ((kr >> 4) << 3);
        bph[((wv * KSTEPS + ks) * 32 + lb) * 16 + e] =
            (_Float16)(pad[u] ? 0.5f : xv[u]);
      }
    }
  }
  __syncthreads();

  // ---- Hot loop: pure LDS b128 loads + WMMA ----
  v8f acc0 = {}, acc1 = {}, acc2 = {}, acc3 = {};
  #pragma unroll
  for (int ks = 0; ks < KSTEPS; ++ks) {
    const v16h bfrag = bp_v[(wave * KSTEPS + ks) * 32 + lane];
    const v16h a0 = aw_v[(ks * 4 + 0) * 32 + lane];
    const v16h a1 = aw_v[(ks * 4 + 1) * 32 + lane];
    const v16h a2 = aw_v[(ks * 4 + 2) * 32 + lane];
    const v16h a3 = aw_v[(ks * 4 + 3) * 32 + lane];
    acc0 = __builtin_amdgcn_wmma_f32_16x16x32_f16(false, a0, false, bfrag, (short)0, acc0, false, false);
    acc1 = __builtin_amdgcn_wmma_f32_16x16x32_f16(false, a1, false, bfrag, (short)0, acc1, false, false);
    acc2 = __builtin_amdgcn_wmma_f32_16x16x32_f16(false, a2, false, bfrag, (short)0, acc2, false, false);
    acc3 = __builtin_amdgcn_wmma_f32_16x16x32_f16(false, a3, false, bfrag, (short)0, acc3, false, false);
  }

  // ---- Store (lanes 0-15 = consecutive j -> coalesced) + stats ----
  const int  half  = lane >> 4;
  const int  l15   = lane & 15;
  const int  half8 = half * 8;
  const long obase = (long)b0 * (COUT * PIX) + (long)(i0 * LSZ + j0 + l15);

  auto emit = [&](const v8f& a, int t) {
    #pragma unroll
    for (int r = 0; r < 8; ++r) {
      const int   ch = t * 16 + r + half8;      // D row -> channel
      const float v  = a[r];
      out[obase + (long)ch * PIX] = v;
      float s = v, q = v * v;
      #pragma unroll
      for (int d = 1; d < 16; d <<= 1) {        // reduce 16-lane half
        s += __shfl_xor(s, d, 32);
        q += __shfl_xor(q, d, 32);
      }
      if (l15 == 0) {
        atomicAdd(&s_sum[ch], s);
        atomicAdd(&s_sq[ch], q);
      }
    }
  };
  emit(acc0, 0); emit(acc1, 1); emit(acc2, 2); emit(acc3, 3);

  __syncthreads();
  if (tid < COUT) {
    atomicAdd(&stats[tid],        s_sum[tid]);
    atomicAdd(&stats[COUT + tid], s_sq[tid]);
  }
}

// ---------------------------------------------------------------------------
// Kernel 2: fold mean/var/gamma/beta into per-channel scale+bias
// ---------------------------------------------------------------------------
__global__ void foveal_stats(const float* __restrict__ sums,
                             const float* __restrict__ gamma,
                             const float* __restrict__ beta,
                             float* __restrict__ sb)
{
  const int o = threadIdx.x;
  const float inv  = 1.0f / (float)MTOT;
  const float mean = sums[o] * inv;
  const float var  = sums[COUT + o] * inv - mean * mean;
  const float sc   = gamma[o] * rsqrtf(var + EPSV);
  sb[o]        = sc;
  sb[COUT + o] = beta[o] - mean * sc;
}

// ---------------------------------------------------------------------------
// Kernel 3: in-place normalize + ReLU, float4 vectorized
// ---------------------------------------------------------------------------
__global__ __launch_bounds__(256) void foveal_norm(float* __restrict__ out,
                                                   const float* __restrict__ sb)
{
  const long base = ((long)blockIdx.x * 256 + threadIdx.x) * 4;
  const int  o    = (int)((base / PIX) & (COUT - 1));
  const float sc = sb[o];
  const float bi = sb[COUT + o];
  float4 v = *reinterpret_cast<float4*>(out + base);
  v.x = fmaxf(fmaf(v.x, sc, bi), 0.f);
  v.y = fmaxf(fmaf(v.y, sc, bi), 0.f);
  v.z = fmaxf(fmaf(v.z, sc, bi), 0.f);
  v.w = fmaxf(fmaf(v.w, sc, bi), 0.f);
  *reinterpret_cast<float4*>(out + base) = v;
}

// ---------------------------------------------------------------------------
extern "C" void kernel_launch(void* const* d_in, const int* in_sizes, int n_in,
                              void* d_out, int out_size, void* d_ws, size_t ws_size,
                              hipStream_t stream)
{
  const float* x     = (const float*)d_in[0];
  const float* w     = (const float*)d_in[1];
  const float* gamma = (const float*)d_in[2];
  const float* beta  = (const float*)d_in[3];
  const int*   li    = (const int*)d_in[4];
  const int*   lj    = (const int*)d_in[5];
  float* out = (float*)d_out;
  float* ws  = (float*)d_ws;           // [0..127] sums, [128..255] scale/bias

  hipMemsetAsync(ws, 0, 2 * COUT * sizeof(float), stream);

  const int mtiles = MTOT / 16;        // 25088
  foveal_gemm<<<mtiles / 8, 256, 0, stream>>>(x, w, li, lj, out, ws);
  foveal_stats<<<1, COUT, 0, stream>>>(ws, gamma, beta, ws + 2 * COUT);
  foveal_norm<<<(long)MTOT * COUT / (256 * 4), 256, 0, stream>>>(out, ws + 2 * COUT);
}
